// TransformerLayer_48515950575666
// MI455X (gfx1250) — matrix operand order repaired and checked by reading, not compiled, not run on hardware
//
#include <hip/hip_runtime.h>
#include <hip/hip_bf16.h>
#include <stdint.h>

// ---------------- problem constants ----------------
#define DM     1024
#define HEADS  16
#define DI     64
#define TSEQ   2048
#define BATCH  4
#define MROWS  (BATCH * TSEQ)   // 8192
#define SQRT_DM 32.0f

typedef unsigned short bf16_t;
typedef __attribute__((ext_vector_type(16))) __bf16 v16bf;
typedef __attribute__((ext_vector_type(8)))  float  v8f;
typedef __attribute__((ext_vector_type(4)))  unsigned int u32x4;
typedef __attribute__((ext_vector_type(8)))  int i32x8;
typedef __attribute__((ext_vector_type(4)))  int i32x4;

#if __has_builtin(__builtin_amdgcn_tensor_load_to_lds) && \
    __has_builtin(__builtin_amdgcn_s_wait_tensorcnt)
#define HAVE_TDM 1
#else
#define HAVE_TDM 0
#endif

// ---------------- small helpers ----------------
__device__ __forceinline__ bf16_t f2bf(float f) {
  unsigned int u = __float_as_uint(f);
  u += 0x7FFFu + ((u >> 16) & 1u);          // round-to-nearest-even
  return (bf16_t)(u >> 16);
}
__device__ __forceinline__ float bf2f(bf16_t h) {
  return __uint_as_float(((unsigned int)h) << 16);
}
__device__ __forceinline__ v8f zero8() {
  v8f z = {0.f, 0.f, 0.f, 0.f, 0.f, 0.f, 0.f, 0.f};
  return z;
}

union FragU { v16bf v; uint4 u[2]; unsigned int w[8]; };

// A-fragment 16x32 bf16 (row-major source, stride lda elems).
// lane = m (0..15) + 16*hi ; elems i<8 -> k = i + 8*hi ; i>=8 -> k = 16+(i-8)+8*hi
__device__ __forceinline__ v16bf load_frag_a(const bf16_t* base, int lda, int lane) {
  const int m = lane & 15, hi = lane >> 4;
  FragU f;
  const bf16_t* r = base + (size_t)m * lda + 8 * hi;
  f.u[0] = *reinterpret_cast<const uint4*>(r);
  f.u[1] = *reinterpret_cast<const uint4*>(r + 16);
  return f.v;
}

// B-fragment 32x16 bf16 where B(k,n) = W[n][k] (W row-major, stride ldb).
// lane = n (0..15) + 16*hi ; elems i<8 -> k = i + 8*hi ; i>=8 -> k = 16+(i-8)+8*hi
__device__ __forceinline__ v16bf load_frag_b(const bf16_t* base, int ldb, int lane) {
  const int n = lane & 15, hi = lane >> 4;
  FragU f;
  const bf16_t* r = base + (size_t)n * ldb + 8 * hi;
  f.u[0] = *reinterpret_cast<const uint4*>(r);
  f.u[1] = *reinterpret_cast<const uint4*>(r + 16);
  return f.v;
}

__device__ __forceinline__ v8f wmma_bf16(v16bf a, v16bf b, v8f c) {
  return __builtin_amdgcn_wmma_f32_16x16x32_bf16(
      /*neg_a=*/false, a, /*neg_b=*/false, b,
      /*c_mod=*/(short)0, c, /*reuse_a=*/false, /*reuse_b=*/false);
}

__device__ __forceinline__ float block_sum(float v, float* smem) {
  #pragma unroll
  for (int o = 1; o < 32; o <<= 1) v += __shfl_xor(v, o, 32);
  const int w = threadIdx.x >> 5;
  if ((threadIdx.x & 31) == 0) smem[w] = v;
  __syncthreads();
  if (threadIdx.x < 32) {
    float t = (threadIdx.x < (int)(blockDim.x >> 5)) ? smem[threadIdx.x] : 0.0f;
    #pragma unroll
    for (int o = 1; o < 8; o <<= 1) t += __shfl_xor(t, o, 32);
    if (threadIdx.x == 0) smem[0] = t;
  }
  __syncthreads();
  v = smem[0];
  __syncthreads();
  return v;
}

#if HAVE_TDM
// Issue one TDM 2D tile load: gptr = tile start (row-major, elem stride0),
// tile is tile_d1 rows x tile_d0 bf16 elems -> contiguous into LDS at lds_off.
__device__ __forceinline__ void tdm_load_2d(const bf16_t* gptr, unsigned lds_off,
                                            int tensor_d0, int tensor_d1,
                                            int tile_d0, int tile_d1, int stride0) {
  const unsigned long long ga = (unsigned long long)(uintptr_t)gptr;
  u32x4 g0;
  g0[0] = 1u;                                              // count=1, user mode
  g0[1] = lds_off;                                         // lds_addr (bytes)
  g0[2] = (unsigned)(ga & 0xFFFFFFFFu);                    // global_addr[31:0]
  g0[3] = (unsigned)((ga >> 32) & 0x01FFFFFFu) | 0x80000000u; // addr[56:32] | type=2
  i32x8 g1;
  g1[0] = 0x00010000;                                      // data_size=1 (2 bytes)
  g1[1] = (int)(((unsigned)tensor_d0 & 0xFFFFu) << 16);    // tensor_dim0[15:0]
  g1[2] = (int)((((unsigned)tensor_d1 & 0xFFFFu) << 16) |  // tensor_dim1[15:0]
                (((unsigned)tensor_d0 >> 16) & 0xFFFFu));  // tensor_dim0[31:16]
  g1[3] = (int)((((unsigned)tile_d0 & 0xFFFFu) << 16) |    // tile_dim0
                (((unsigned)tensor_d1 >> 16) & 0xFFFFu));  // tensor_dim1[31:16]
  g1[4] = (int)((unsigned)tile_d1 & 0xFFFFu);              // tile_dim1 (tile_dim2=0)
  g1[5] = stride0;                                         // tensor_dim0_stride[31:0]
  g1[6] = 0;
  g1[7] = 0;
  const i32x4 z4 = {0, 0, 0, 0};
#if defined(__clang_major__) && (__clang_major__ >= 23)
  const i32x8 z8 = {0, 0, 0, 0, 0, 0, 0, 0};
  __builtin_amdgcn_tensor_load_to_lds(g0, g1, z4, z4, z8, 0);
#else
  __builtin_amdgcn_tensor_load_to_lds(g0, g1, z4, z4, 0);
#endif
}
#endif

// ---------------- kernels ----------------

// fp32 -> bf16, 4 elems / thread
__global__ __launch_bounds__(256) void cvt_kernel(const float* __restrict__ in,
                                                  bf16_t* __restrict__ out, int n4) {
  const int i = blockIdx.x * 256 + threadIdx.x;
  if (i < n4) {
    const float4 f = reinterpret_cast<const float4*>(in)[i];
    union { bf16_t h[4]; unsigned long long u; } o;
    o.h[0] = f2bf(f.x); o.h[1] = f2bf(f.y); o.h[2] = f2bf(f.z); o.h[3] = f2bf(f.w);
    reinterpret_cast<unsigned long long*>(out)[i] = o.u;
  }
}

// per-head score upper bound: |Q.K| <= max_d(sqk_eff)^2, score = sqrt_dm * that
__global__ void bound_kernel(const float* __restrict__ sqk, float* __restrict__ bound) {
  const int h = threadIdx.x;
  if (h < HEADS) {
    float m = 0.f;
    for (int d = 0; d < DI; ++d) m = fmaxf(m, fabsf(sqk[h * DI + d]) * SQRT_DM);
    bound[h] = SQRT_DM * m * m;
  }
}

// Y[M,N] = A[M,K] @ W[N,K]^T  (bf16 operands, fp32 accumulate).
// block = 128 (4 waves); wave computes 16(M) x 64(N); grid (N/64, M/64).
// B tile (64 x 64, 8KB) double-buffered in LDS via Tensor Data Mover:
// issue tile kk+1, wait tensorcnt<=1 (tile kk done), compute 8 WMMAs, barrier.
template <bool OUT_BF16>
__global__ __launch_bounds__(128) void gemm_kernel(const bf16_t* __restrict__ A,
                                                   const bf16_t* __restrict__ W,
                                                   void* __restrict__ Y,
                                                   int M, int N, int K) {
#if HAVE_TDM
  __shared__ __align__(16) bf16_t tileB[2][64][64];
#endif
  const int lane  = threadIdx.x & 31;
  const int wave  = threadIdx.x >> 5;
  const int mBase = blockIdx.y * 64 + wave * 16;
  const int nBase = blockIdx.x * 64;
  const bf16_t* Arow = A + (size_t)mBase * K;
  v8f acc[4] = {zero8(), zero8(), zero8(), zero8()};
  const int nk = K >> 6;                       // 64-wide k steps
#if HAVE_TDM
  if (wave == 0)                               // prologue: tile 0 in flight
    tdm_load_2d(W + (size_t)nBase * K, (unsigned)(uintptr_t)&tileB[0][0][0],
                K, N, 64, 64, K);
#endif
  for (int kk = 0; kk < nk; ++kk) {
    const int k = kk << 6;
#if HAVE_TDM
    const int cur = kk & 1;
    if (wave == 0) {
      if (kk + 1 < nk) {                       // overlap next DMA with compute
        tdm_load_2d(W + (size_t)nBase * K + (k + 64),
                    (unsigned)(uintptr_t)&tileB[cur ^ 1][0][0], K, N, 64, 64, K);
        __builtin_amdgcn_s_wait_tensorcnt(1);  // tile kk complete, kk+1 in flight
      } else {
        __builtin_amdgcn_s_wait_tensorcnt(0);
      }
    }
    __syncthreads();
#endif
    #pragma unroll
    for (int kc = 0; kc < 2; ++kc) {
      if (k + 512 < K)
        __builtin_prefetch(Arow + (size_t)(lane & 15) * K + k + 512, 0, 1);
      const v16bf af = load_frag_a(Arow + k + kc * 32, K, lane);
      #pragma unroll
      for (int j = 0; j < 4; ++j) {
#if HAVE_TDM
        const v16bf bfr = load_frag_b(&tileB[cur][16 * j][kc * 32], 64, lane);
#else
        const v16bf bfr =
            load_frag_b(W + (size_t)(nBase + 16 * j) * K + k + kc * 32, K, lane);
#endif
        acc[j] = wmma_bf16(af, bfr, acc[j]);
      }
    }
#if HAVE_TDM
    __syncthreads();                           // tile cur free before overwrite
#endif
  }
  const int lo = lane & 15, hi = lane >> 4;
  #pragma unroll
  for (int j = 0; j < 4; ++j) {
    const int col = nBase + 16 * j + lo;
    #pragma unroll
    for (int r = 0; r < 8; ++r) {
      const int row = mBase + r + 8 * hi;
      if (OUT_BF16)
        ((bf16_t*)Y)[(size_t)row * N + col] = f2bf(acc[j][r]);
      else
        ((float*)Y)[(size_t)row * N + col] = acc[j][r];
    }
  }
}

// RoPE + justnorm + sqk scaling; V written transposed.  One wave per (b,h,t).
__global__ __launch_bounds__(256) void rope_kernel(const bf16_t* __restrict__ QKV,
                                                   const float* __restrict__ sqk,
                                                   bf16_t* __restrict__ Qn,
                                                   bf16_t* __restrict__ Kn,
                                                   bf16_t* __restrict__ Vt) {
  const int lane = threadIdx.x & 31;
  const int gw   = blockIdx.x * 8 + (threadIdx.x >> 5);
  const int t    = gw % TSEQ;
  const int bh   = gw / TSEQ;                 // b*HEADS + h
  if (bh >= BATCH * HEADS) return;
  const int b = bh >> 4, h = bh & 15;
  const bf16_t* row = QKV + ((size_t)(b * TSEQ + t)) * (3 * HEADS * DI) + h * (3 * DI);
  const float q0 = bf2f(row[lane]),       q1 = bf2f(row[lane + 32]);
  const float k0 = bf2f(row[64 + lane]),  k1 = bf2f(row[96 + lane]);
  const float v0 = bf2f(row[128 + lane]), v1 = bf2f(row[160 + lane]);
  const float ang = (float)t * __expf(-9.2103403720f * ((float)lane) * (1.0f / 32.0f));
  float sn, cs;
  __sincosf(ang, &sn, &cs);
  const float qr0 = q0 * cs - q1 * sn, qr1 = q1 * cs + q0 * sn;
  const float kr0 = k0 * cs - k1 * sn, kr1 = k1 * cs + k0 * sn;
  float ssq = qr0 * qr0 + qr1 * qr1, ssk = kr0 * kr0 + kr1 * kr1;
  #pragma unroll
  for (int o = 1; o < 32; o <<= 1) {
    ssq += __shfl_xor(ssq, o, 32);
    ssk += __shfl_xor(ssk, o, 32);
  }
  const float rq = rsqrtf(ssq), rk = rsqrtf(ssk);
  const float s0 = sqk[h * DI + lane] * SQRT_DM;
  const float s1 = sqk[h * DI + lane + 32] * SQRT_DM;
  bf16_t* qo = Qn + ((size_t)bh * TSEQ + t) * DI;
  bf16_t* ko = Kn + ((size_t)bh * TSEQ + t) * DI;
  qo[lane] = f2bf(qr0 * rq * s0); qo[lane + 32] = f2bf(qr1 * rq * s1);
  ko[lane] = f2bf(kr0 * rk * s0); ko[lane + 32] = f2bf(kr1 * rk * s1);
  Vt[((size_t)bh * DI + lane) * TSEQ + t]      = f2bf(v0);
  Vt[((size_t)bh * DI + lane + 32) * TSEQ + t] = f2bf(v1);
}

// Causal flash attention, one wave per 16-row Q tile. grid (T/64, B*H), block 128.
// Static per-head max bound -> no online rescaling; row sums via WMMA with ones-B.
__global__ __launch_bounds__(128) void attn_kernel(const bf16_t* __restrict__ Qn,
                                                   const bf16_t* __restrict__ Kn,
                                                   const bf16_t* __restrict__ Vt,
                                                   const float* __restrict__ Bound,
                                                   bf16_t* __restrict__ Out) {
  __shared__ __align__(16) bf16_t pbuf[4][16][40];   // per-wave P tile, padded rows
  const int lane = threadIdx.x & 31;
  const int wave = threadIdx.x >> 5;
  const int lo = lane & 15, hi = lane >> 4;
  const int bh = blockIdx.y;
  const int qBase = (blockIdx.x * 4 + wave) * 16;
  const float mx = Bound[bh & 15];
  const bf16_t* Qb = Qn + ((size_t)bh * TSEQ + qBase) * DI;
  const v16bf qa0 = load_frag_a(Qb, DI, lane);        // k = 0..31
  const v16bf qa1 = load_frag_a(Qb + 32, DI, lane);   // k = 32..63
  FragU onesf;                                        // all-ones bf16 B matrix
  #pragma unroll
  for (int i = 0; i < 8; ++i) onesf.w[i] = 0x3F803F80u;
  v8f o0 = zero8(), o1 = zero8(), o2 = zero8(), o3 = zero8(), osum = zero8();
  const int kvEnd = qBase + 16;
  for (int kv = 0; kv < kvEnd; kv += 32) {
    #pragma unroll
    for (int sub = 0; sub < 2; ++sub) {
      const int n0 = kv + sub * 16;
      v8f c = zero8();
      if (n0 < kvEnd) {
        const bf16_t* Kb = Kn + ((size_t)bh * TSEQ + n0) * DI;
        c = wmma_bf16(qa0, load_frag_b(Kb, DI, lane), c);
        c = wmma_bf16(qa1, load_frag_b(Kb + 32, DI, lane), c);
      }
      const int col = n0 + lo;
      #pragma unroll
      for (int r = 0; r < 8; ++r) {
        const int rowg = qBase + r + 8 * hi;
        const float s = (n0 < kvEnd && col <= rowg) ? c[r] * SQRT_DM : -1e30f;
        pbuf[wave][r + 8 * hi][sub * 16 + lo] = f2bf(__expf(s - mx));
      }
    }
    // intra-wave LDS transpose fence: P stores must land before A-frag reads
    asm volatile("s_wait_dscnt 0" ::: "memory");
    FragU pf;
    pf.u[0] = *reinterpret_cast<const uint4*>(&pbuf[wave][lo][8 * hi]);
    pf.u[1] = *reinterpret_cast<const uint4*>(&pbuf[wave][lo][16 + 8 * hi]);
    const bf16_t* Vb = Vt + (size_t)bh * DI * TSEQ + kv;
    o0 = wmma_bf16(pf.v, load_frag_b(Vb,             TSEQ, lane), o0);
    o1 = wmma_bf16(pf.v, load_frag_b(Vb + 16 * TSEQ, TSEQ, lane), o1);
    o2 = wmma_bf16(pf.v, load_frag_b(Vb + 32 * TSEQ, TSEQ, lane), o2);
    o3 = wmma_bf16(pf.v, load_frag_b(Vb + 48 * TSEQ, TSEQ, lane), o3);
    osum = wmma_bf16(pf.v, onesf.v, osum);          // row sums, no shuffles
  }
  const int bb = bh >> 4, hh = bh & 15;
  #pragma unroll
  for (int r = 0; r < 8; ++r) {
    const float inv = 1.0f / osum[r];
    const int t = qBase + r + 8 * hi;
    bf16_t* orow = Out + ((size_t)(bb * TSEQ + t)) * DM + hh * DI;
    orow[lo]      = f2bf(o0[r] * inv);
    orow[16 + lo] = f2bf(o1[r] * inv);
    orow[32 + lo] = f2bf(o2[r] * inv);
    orow[48 + lo] = f2bf(o3[r] * inv);
  }
}

// x1 = justnorm(A + lr*(Bn - A)); A=justnorm(X) row, Bn=justnorm(Tt) row
__global__ __launch_bounds__(256) void residual_kernel(const float* __restrict__ X,
                                                       const float* __restrict__ Tt,
                                                       const float* __restrict__ alpha,
                                                       float coef,
                                                       float* __restrict__ Xout,
                                                       bf16_t* __restrict__ Xbf) {
  __shared__ float smem[8];
  const int row = blockIdx.x, tid = threadIdx.x;
  const float* xr = X + (size_t)row * DM;
  const float* tr = Tt + (size_t)row * DM;
  float xv[4], tv[4], ssx = 0.f, sst = 0.f;
  #pragma unroll
  for (int i = 0; i < 4; ++i) {
    const int idx = tid + 256 * i;
    xv[i] = xr[idx]; tv[i] = tr[idx];
    ssx += xv[i] * xv[i]; sst += tv[i] * tv[i];
  }
  ssx = block_sum(ssx, smem);
  sst = block_sum(sst, smem);
  const float rx = rsqrtf(ssx), rt = rsqrtf(sst);
  float c[4], ssc = 0.f;
  #pragma unroll
  for (int i = 0; i < 4; ++i) {
    const int idx = tid + 256 * i;
    const float lr = fabsf(alpha[idx] * coef);
    const float a = xv[i] * rx, bn = tv[i] * rt;
    c[i] = a + lr * (bn - a);
    ssc += c[i] * c[i];
  }
  ssc = block_sum(ssc, smem);
  const float rc = rsqrtf(ssc);
  #pragma unroll
  for (int i = 0; i < 4; ++i) {
    const int idx = tid + 256 * i;
    const float o = c[i] * rc;
    Xout[(size_t)row * DM + idx] = o;
    if (Xbf) Xbf[(size_t)row * DM + idx] = f2bf(o);
  }
}

// tn = justnorm(suv*sqrt_dm*uv) over 8192; res = u * silu(v)
__global__ __launch_bounds__(256) void mlp_act_kernel(const bf16_t* __restrict__ UV,
                                                      const float* __restrict__ suv,
                                                      bf16_t* __restrict__ Res) {
  __shared__ float smem[8];
  const int row = blockIdx.x, tid = threadIdx.x;
  const bf16_t* r = UV + (size_t)row * (8 * DM);
  float ss = 0.f;
  #pragma unroll
  for (int i = 0; i < 32; ++i) {
    const int idx = tid + 256 * i;
    const float u = bf2f(r[idx]) * (suv[idx] * SQRT_DM);
    ss += u * u;
  }
  ss = block_sum(ss, smem);
  const float rn = rsqrtf(ss);
  #pragma unroll
  for (int i = 0; i < 16; ++i) {
    const int iu = tid + 256 * i;
    const int iv = iu + 4 * DM;
    const float u = bf2f(r[iu]) * (suv[iu] * SQRT_DM) * rn;
    const float v = bf2f(r[iv]) * (suv[iv] * SQRT_DM) * rn;
    const float sv = v * (1.0f / (1.0f + __expf(-v)));
    Res[(size_t)row * (4 * DM) + iu] = f2bf(u * sv);
  }
}

// ---------------- workspace layout (bytes) ----------------
static constexpr size_t OFF_XB    = 0;                                  // bf16 X  [8192,1024]
static constexpr size_t SZ_XB     = (size_t)MROWS * DM * 2;
static constexpr size_t OFF_WOB   = OFF_XB + SZ_XB;                     // bf16 W_O [1024,1024]
static constexpr size_t SZ_WOB    = (size_t)DM * DM * 2;
static constexpr size_t OFF_WUB   = OFF_WOB + SZ_WOB;                   // bf16 Wu [8192,1024]
static constexpr size_t SZ_WUB    = (size_t)8 * DM * DM * 2;
static constexpr size_t OFF_WVB   = OFF_WUB + SZ_WUB;                   // bf16 Wv [1024,4096]
static constexpr size_t SZ_WVB    = (size_t)DM * 4 * DM * 2;
static constexpr size_t OFF_X1    = OFF_WVB + SZ_WVB;                   // f32 x1 [8192,1024]
static constexpr size_t SZ_X1     = (size_t)MROWS * DM * 4;
static constexpr size_t OFF_X1B   = OFF_X1 + SZ_X1;                     // bf16 x1
static constexpr size_t SZ_X1B    = (size_t)MROWS * DM * 2;
static constexpr size_t OFF_RES   = OFF_X1B + SZ_X1B;                   // bf16 res [8192,4096]
static constexpr size_t SZ_RES    = (size_t)MROWS * 4 * DM * 2;
static constexpr size_t OFF_TMLP  = OFF_RES + SZ_RES;                   // f32 mlp out [8192,1024]
static constexpr size_t SZ_TMLP   = (size_t)MROWS * DM * 4;
// transient region (phase 1: qkv/attn; phase 2 aliases it for UV)
static constexpr size_t OFF_TR    = OFF_TMLP + SZ_TMLP;
static constexpr size_t OFF_WQKVB = OFF_TR;                             // bf16 [3072,1024]
static constexpr size_t SZ_WQKVB  = (size_t)3 * HEADS * DI * DM * 2;
static constexpr size_t OFF_QKV   = OFF_WQKVB + SZ_WQKVB;               // bf16 [8192,3072]
static constexpr size_t SZ_QKV    = (size_t)MROWS * 3 * HEADS * DI * 2;
static constexpr size_t OFF_QN    = OFF_QKV + SZ_QKV;                   // bf16 [64,2048,64]
static constexpr size_t SZ_QN     = (size_t)BATCH * HEADS * TSEQ * DI * 2;
static constexpr size_t OFF_KN    = OFF_QN + SZ_QN;
static constexpr size_t OFF_VT    = OFF_KN + SZ_QN;                     // bf16 [64,64,2048]
static constexpr size_t OFF_AOUT  = OFF_VT + SZ_QN;                     // bf16 [8192,1024]
static constexpr size_t OFF_TPROJ = OFF_AOUT + SZ_X1B;                  // f32 [8192,1024]
static constexpr size_t OFF_BOUND = OFF_TPROJ + SZ_TMLP;                // f32 [16]
static constexpr size_t OFF_UV    = OFF_TR;                             // bf16 [8192,8192] (alias)

extern "C" void kernel_launch(void* const* d_in, const int* in_sizes, int n_in,
                              void* d_out, int out_size, void* d_ws, size_t ws_size,
                              hipStream_t stream) {
  (void)in_sizes; (void)n_in; (void)out_size; (void)ws_size;
  const float* x     = (const float*)d_in[0];
  const float* qkvw  = (const float*)d_in[1];
  const float* sqk   = (const float*)d_in[2];
  const float* W_O   = (const float*)d_in[3];
  const float* Wu    = (const float*)d_in[4];
  const float* Wv    = (const float*)d_in[5];
  const float* a_att = (const float*)d_in[6];
  const float* a_mlp = (const float*)d_in[7];
  const float* suv   = (const float*)d_in[8];

  char* ws = (char*)d_ws;
  bf16_t* XB    = (bf16_t*)(ws + OFF_XB);
  bf16_t* WOB   = (bf16_t*)(ws + OFF_WOB);
  bf16_t* WUB   = (bf16_t*)(ws + OFF_WUB);
  bf16_t* WVB   = (bf16_t*)(ws + OFF_WVB);
  float*  X1    = (float*)(ws + OFF_X1);
  bf16_t* X1B   = (bf16_t*)(ws + OFF_X1B);
  bf16_t* RES   = (bf16_t*)(ws + OFF_RES);
  float*  TMLP  = (float*)(ws + OFF_TMLP);
  bf16_t* WQKVB = (bf16_t*)(ws + OFF_WQKVB);
  bf16_t* QKV   = (bf16_t*)(ws + OFF_QKV);
  bf16_t* QN    = (bf16_t*)(ws + OFF_QN);
  bf16_t* KN    = (bf16_t*)(ws + OFF_KN);
  bf16_t* VT    = (bf16_t*)(ws + OFF_VT);
  bf16_t* AOUT  = (bf16_t*)(ws + OFF_AOUT);
  float*  TPROJ = (float*)(ws + OFF_TPROJ);
  float*  BOUND = (float*)(ws + OFF_BOUND);
  bf16_t* UV    = (bf16_t*)(ws + OFF_UV);

  // 1) fp32 -> bf16 conversions + per-head score bound
  cvt_kernel<<<(MROWS * DM / 4) / 256, 256, 0, stream>>>(x, XB, MROWS * DM / 4);
  cvt_kernel<<<(3 * HEADS * DI * DM / 4) / 256, 256, 0, stream>>>(qkvw, WQKVB, 3 * HEADS * DI * DM / 4);
  cvt_kernel<<<(DM * DM / 4) / 256, 256, 0, stream>>>(W_O, WOB, DM * DM / 4);
  cvt_kernel<<<(8 * DM * DM / 4) / 256, 256, 0, stream>>>(Wu, WUB, 8 * DM * DM / 4);
  cvt_kernel<<<(4 * DM * DM / 4) / 256, 256, 0, stream>>>(Wv, WVB, 4 * DM * DM / 4);
  bound_kernel<<<1, 32, 0, stream>>>(sqk, BOUND);

  // 2) QKV projection: [8192,1024] @ [3072,1024]^T
  gemm_kernel<true><<<dim3(3072 / 64, MROWS / 64), 128, 0, stream>>>(XB, WQKVB, QKV, MROWS, 3072, DM);

  // 3) RoPE + norm + sqk; V transposed
  rope_kernel<<<(BATCH * HEADS * TSEQ) / 8, 256, 0, stream>>>(QKV, sqk, QN, KN, VT);

  // 4) causal flash attention (static-max softmax, WMMA row sums)
  attn_kernel<<<dim3(TSEQ / 64, BATCH * HEADS), 128, 0, stream>>>(QN, KN, VT, BOUND, AOUT);

  // 5) output projection W_O
  gemm_kernel<false><<<dim3(DM / 64, MROWS / 64), 128, 0, stream>>>(AOUT, WOB, TPROJ, MROWS, DM, DM);

  // 6) attention residual lerp + norm
  residual_kernel<<<MROWS, 256, 0, stream>>>(x, TPROJ, a_att, 0.05f * SQRT_DM, X1, X1B);

  // 7) MLP up: [8192,1024] @ [8192,1024]^T
  gemm_kernel<true><<<dim3(8 * DM / 64, MROWS / 64), 128, 0, stream>>>(X1B, WUB, UV, MROWS, 8 * DM, DM);

  // 8) norm over 8192 + SwiGLU
  mlp_act_kernel<<<MROWS, 256, 0, stream>>>(UV, suv, RES);

  // 9) MLP down: [8192,4096] @ [1024,4096]^T
  gemm_kernel<false><<<dim3(DM / 64, MROWS / 64), 128, 0, stream>>>(RES, WVB, TMLP, MROWS, DM, 4 * DM);

  // 10) MLP residual lerp + norm -> d_out (fp32)
  residual_kernel<<<MROWS, 256, 0, stream>>>(X1, TMLP, a_mlp, 0.05f, (float*)d_out, nullptr);
}
